// HyperGNNTernaryGraph_15951508537729
// MI455X (gfx1250) — matrix operand order repaired and checked by reading, not compiled, not run on hardware
//
#include <hip/hip_runtime.h>

typedef __attribute__((ext_vector_type(16))) _Float16 v16h;
typedef __attribute__((ext_vector_type(8)))  _Float16 v8h;
typedef __attribute__((ext_vector_type(8)))  float    v8f;

constexpr int    kD     = 256;
constexpr int    kNE    = 64;
constexpr int    kBS    = 16;
constexpr int    ROWS_E = kBS * kNE;             // 1024 entity rows
constexpr int    ROWS_R = kBS * kNE * kNE;       // 65536 edge rows
constexpr size_t SZ_E   = (size_t)ROWS_E * kD;   // 262144 floats
constexpr size_t SZ_R   = (size_t)ROWS_R * kD;   // 16777216 floats
constexpr size_t SZ_W   = (size_t)kD * kD;       // 65536 elements per weight
constexpr float  kNEG   = -10000.0f;

__device__ __forceinline__ float gelu_f(float x) {
  // exact-erf GELU, matching torch nn.GELU default
  return 0.5f * x * (1.0f + erff(x * 0.70710678118654752440f));
}
__device__ __forceinline__ float wave_sum(float s) {
#pragma unroll
  for (int o = 16; o > 0; o >>= 1) s += __shfl_xor(s, o, 32);
  return s;
}

// ---------------------------------------------------------------------------
// Weight pre-pass: fp32 [K][N] -> fp16 transposed [N][K], once per launch.
// Makes every WMMA B-fragment a single contiguous 32B (v16h) load.
// ---------------------------------------------------------------------------
__global__ __launch_bounds__(256) void k_cvt_wt(const float* __restrict__ W,
                                                _Float16* __restrict__ Wt)
{
  const int idx = blockIdx.x * 256 + threadIdx.x;      // 0 .. 65535
  const int k = idx >> 8, n = idx & 255;
  Wt[(size_t)n * kD + k] = (_Float16)W[idx];
}

// ---------------------------------------------------------------------------
// WMMA GEMM: C[M,256] = A[M,256] @ W[256,256] (+ bias) (+ resid)
// block = 256 threads (8 waves); block tile = 32(M) x 256(N) over K=256.
// A tile (32x256) staged ONCE in LDS as fp16 (float4 global loads), shared by
// all 8 waves; wave = (m half-strip, 64-wide N group) with 4 16x16 acc tiles,
// so each A fragment feeds 4 v_wmma_f32_16x16x32_f16 issues per k-step.
// Wt is fp16 [N][K] so B fragments are single v16h (b128) loads.
// A and B fragments both use contiguous k order: k = 16*(lane/16) + element.
// ---------------------------------------------------------------------------
__global__ __launch_bounds__(256) void k_gemm256(
    const float* __restrict__ A, const _Float16* __restrict__ Wt,
    const float* __restrict__ bias, const float* __restrict__ resid,
    float* __restrict__ C, int M)
{
  __shared__ _Float16 As[32 * 256];
  const int tid  = threadIdx.x;
  const int lane = tid & 31;
  const int wv   = tid >> 5;          // 0..7
  const int wr   = wv >> 2;           // 0..1 : m sub-strip (16 rows)
  const int wc   = wv & 3;            // 0..3 : 64-wide n group
  const int m0   = blockIdx.x * 32;
  (void)M;

  // stage 32x256 fp32 -> fp16 via float4 loads (2048 float4, 8 per thread)
  for (int t = tid; t < 32 * 64; t += 256) {
    const int r  = t >> 6;            // row 0..31
    const int c4 = (t & 63) << 2;     // col 0,4,..,252
    const float4 f = *(const float4*)(A + (size_t)(m0 + r) * kD + c4);
    _Float16* dst = As + r * 256 + c4;
    dst[0] = (_Float16)f.x; dst[1] = (_Float16)f.y;
    dst[2] = (_Float16)f.z; dst[3] = (_Float16)f.w;
  }
  __syncthreads();

  const int lhi   = (lane >= 16) ? 1 : 0;
  const int lm    = lane & 15;
  const int mBase = wr * 16;

  v8f acc[4] = {v8f{}, v8f{}, v8f{}, v8f{}};
#pragma unroll
  for (int k0 = 0; k0 < 256; k0 += 32) {
    // A 16x32 fp16 fragment: lanes0-15 K=0..15, lanes16-31 K=16..31
    v16h a;
    {
      const _Float16* ap = As + (mBase + lm) * 256 + k0 + lhi * 16;
      const v8h lo = *(const v8h*)(ap);
      const v8h hi = *(const v8h*)(ap + 8);
#pragma unroll
      for (int h = 0; h < 8; ++h) { a[h] = lo[h]; a[h + 8] = hi[h]; }
    }
    const int kb = k0 + lhi * 16;     // B: lanes0-15 K=k0..k0+15, lanes16-31 +16
#pragma unroll
    for (int t = 0; t < 4; ++t) {
      const int n = wc * 64 + t * 16 + lm;
      const v16h b = *(const v16h*)(Wt + (size_t)n * kD + kb);
      acc[t] = __builtin_amdgcn_wmma_f32_16x16x32_f16(false, a, false, b,
                                                      (short)0, acc[t],
                                                      false, false);
    }
  }

#pragma unroll
  for (int t = 0; t < 4; ++t) {
    const int ncol = wc * 64 + t * 16 + lm;
    const float bval = bias ? bias[ncol] : 0.0f;
#pragma unroll
    for (int r = 0; r < 8; ++r) {
      const int mrow = m0 + mBase + r + (lhi ? 8 : 0);   // C VGPR r: M=r / 8+r
      float val = acc[t][r] + bval;
      const size_t off = (size_t)mrow * kD + ncol;
      if (resid) val += resid[off];
      C[off] = val;
    }
  }
}

// ---------------------------------------------------------------------------
// LayerNorm over 256-wide rows (eps=1e-6, identity affine). Wave per row.
// ---------------------------------------------------------------------------
__global__ __launch_bounds__(256) void k_ln(const float* __restrict__ X,
                                            float* __restrict__ Y, int M)
{
  const int row = blockIdx.x * 8 + (threadIdx.x >> 5);
  if (row >= M) return;
  const int lane = threadIdx.x & 31;
  const float* x = X + (size_t)row * kD;
  float v[8], s = 0.f;
#pragma unroll
  for (int c = 0; c < 8; ++c) { v[c] = x[lane + 32 * c]; s += v[c]; }
  s = wave_sum(s);
  const float mean = s * (1.0f / 256.0f);
  float q = 0.f;
#pragma unroll
  for (int c = 0; c < 8; ++c) { const float d = v[c] - mean; q += d * d; }
  q = wave_sum(q);
  const float inv = rsqrtf(q * (1.0f / 256.0f) + 1e-6f);
  float* y = Y + (size_t)row * kD;
#pragma unroll
  for (int c = 0; c < 8; ++c) y[lane + 32 * c] = (v[c] - mean) * inv;
}

// prod[b,i,j,d] = l1[b,i,d] * l2[b,j,d] * l3[b,i,j,d]
__global__ __launch_bounds__(256) void k_prod3(const float* __restrict__ L1,
                                               const float* __restrict__ L2,
                                               const float* __restrict__ L3,
                                               float* __restrict__ Out)
{
  const size_t idx = (size_t)blockIdx.x * 256 + threadIdx.x;
  const int    d = idx & 255;
  const size_t j = (idx >> 8) & 63;
  const size_t i = (idx >> 14) & 63;
  const size_t b = idx >> 20;
  Out[idx] = L1[(b * kNE + i) * kD + d] * L2[(b * kNE + j) * kD + d] * L3[idx];
}

// energy_sub[b,i,j] = sum_d gelu(q[b,i,d] + th[b,i,j,d]) * sv[d], j=0 -> self
__global__ __launch_bounds__(256) void k_energy_sub(
    const float* __restrict__ Q, const float* __restrict__ HW,
    const float* __restrict__ FW, const float* __restrict__ vvec,
    float* __restrict__ E)
{
  const int row = blockIdx.x * 8 + (threadIdx.x >> 5);
  if (row >= kBS * kNE * (kNE + 1)) return;
  const int lane = threadIdx.x & 31;
  const int j  = row % (kNE + 1);
  const int bi = row / (kNE + 1);
  const float* q = Q + (size_t)bi * kD;
  const float* t = (j == 0) ? (HW + (size_t)bi * kD)
                            : (FW + ((size_t)bi * kNE + (j - 1)) * kD);
  float s = 0.f;
#pragma unroll
  for (int c = 0; c < 8; ++c) {
    const int dd = lane + 32 * c;
    s += gelu_f(q[dd] + t[dd]) * vvec[dd];
  }
  s = wave_sum(s);
  if (lane == 0) E[row] = s;
}

// energy_obj[b,jrow,i]; jrow=0 -> self row
__global__ __launch_bounds__(256) void k_energy_obj(
    const float* __restrict__ Q, const float* __restrict__ HW,
    const float* __restrict__ FW, const float* __restrict__ vvec,
    float* __restrict__ E)
{
  const int row = blockIdx.x * 8 + (threadIdx.x >> 5);
  if (row >= kBS * (kNE + 1) * kNE) return;
  const int lane = threadIdx.x & 31;
  const int i  = row & 63;
  const int tt = row >> 6;               // b*(NE+1)+jrow
  const int jrow = tt % (kNE + 1);
  const int b    = tt / (kNE + 1);
  const float* q = Q + ((size_t)b * kNE + i) * kD;
  const float* t = (jrow == 0)
      ? (HW + ((size_t)b * kNE + i) * kD)
      : (FW + (((size_t)b * kNE + (jrow - 1)) * kNE + i) * kD);
  float s = 0.f;
#pragma unroll
  for (int c = 0; c < 8; ++c) {
    const int dd = lane + 32 * c;
    s += gelu_f(q[dd] + t[dd]) * vvec[dd];
  }
  s = wave_sum(s);
  if (lane == 0) E[row] = s;
}

// masked softmax over 65 + weighted context, block per (b,i) subject
__global__ __launch_bounds__(256) void k_softctx_sub(
    const float* __restrict__ E, const float* __restrict__ HT,
    const float* __restrict__ F, const int* __restrict__ ent,
    float* __restrict__ CTX)
{
  __shared__ float sE[kNE + 1];
  __shared__ float sMx, sSum;
  const int bi = blockIdx.x;
  const int b = bi >> 6, i = bi & 63;
  const int d = threadIdx.x;
  const int n = ent[b];
  if (d < kNE + 1) {
    const float val = E[(size_t)bi * (kNE + 1) + d];
    bool masked = false;
    if (d > 0) {
      const int j = d - 1;
      masked = (j == i) || !((i < n) && (j < n));
    }
    sE[d] = masked ? kNEG : val;
  }
  __syncthreads();
  if (d == 0) {
    float mx = -1e30f;
    for (int j = 0; j <= kNE; ++j) mx = fmaxf(mx, sE[j]);
    float s = 0.f;
    for (int j = 0; j <= kNE; ++j) s += expf(sE[j] - mx);
    sMx = mx; sSum = s;
  }
  __syncthreads();
  const float mx = sMx, inv = 1.0f / sSum;
  float acc = expf(sE[0] - mx) * inv * HT[(size_t)bi * kD + d];
  for (int j = 0; j < kNE; ++j)
    acc += expf(sE[j + 1] - mx) * inv * F[((size_t)bi * kNE + j) * kD + d];
  CTX[(size_t)bi * kD + d] = acc;
}

// masked softmax over 65 rows (axis=1) + context, block per (b,i) object
__global__ __launch_bounds__(256) void k_softctx_obj(
    const float* __restrict__ E, const float* __restrict__ HT,
    const float* __restrict__ F, const int* __restrict__ ent,
    float* __restrict__ CTX)
{
  __shared__ float sE[kNE + 1];
  __shared__ float sMx, sSum;
  const int bi = blockIdx.x;
  const int b = bi >> 6, i = bi & 63;
  const int d = threadIdx.x;
  const int n = ent[b];
  if (d < kNE + 1) {
    const float val = E[((size_t)b * (kNE + 1) + d) * kNE + i];
    bool masked = false;
    if (d > 0) {
      const int s = d - 1;
      masked = (s == i) || !((s < n) && (i < n));
    }
    sE[d] = masked ? kNEG : val;
  }
  __syncthreads();
  if (d == 0) {
    float mx = -1e30f;
    for (int j = 0; j <= kNE; ++j) mx = fmaxf(mx, sE[j]);
    float s = 0.f;
    for (int j = 0; j <= kNE; ++j) s += expf(sE[j] - mx);
    sMx = mx; sSum = s;
  }
  __syncthreads();
  const float mx = sMx, inv = 1.0f / sSum;
  float acc = expf(sE[0] - mx) * inv * HT[(size_t)bi * kD + d];
  for (int s = 0; s < kNE; ++s)
    acc += expf(sE[s + 1] - mx) * inv *
           F[(((size_t)b * kNE + s) * kNE + i) * kD + d];
  CTX[(size_t)bi * kD + d] = acc;
}

// per-edge 2-way attention: ctx = p0*htr + p1*factor. CTX may alias G0.
__global__ __launch_bounds__(256) void k_rel_ctx(
    const float* AW, const float* G0, const float* G1,
    const float* __restrict__ HTR, const float* __restrict__ F,
    const float* __restrict__ rv, float* CTX)
{
  const int row = blockIdx.x * 8 + (threadIdx.x >> 5);
  if (row >= ROWS_R) return;
  const int lane = threadIdx.x & 31;
  const size_t base = (size_t)row * kD;
  float s0 = 0.f, s1 = 0.f;
#pragma unroll
  for (int c = 0; c < 8; ++c) {
    const int dd = lane + 32 * c;
    const float aw = AW[base + dd];
    const float rw = rv[dd];
    s0 += gelu_f(aw + G0[base + dd]) * rw;
    s1 += gelu_f(aw + G1[base + dd]) * rw;
  }
  s0 = wave_sum(s0);
  s1 = wave_sum(s1);
  const float mx = fmaxf(s0, s1);
  const float e0 = expf(s0 - mx), e1 = expf(s1 - mx);
  const float p0 = e0 / (e0 + e1), p1 = e1 / (e0 + e1);
#pragma unroll
  for (int c = 0; c < 8; ++c) {
    const int dd = lane + 32 * c;
    CTX[base + dd] = p0 * HTR[base + dd] + p1 * F[base + dd];
  }
}

// ---------------------------------------------------------------------------
// Host orchestration
// ---------------------------------------------------------------------------
struct P {
  const int* ent;
  // fp32 biases / vectors
  const float *b1,*b2,*b3,*be,*bps,*bpo,*bpr,*bas,*bao,*bar,*sv,*ov,*rv,
              *bfs,*bfo,*bfr;
  // fp16 transposed [N][K] weights (in workspace)
  const _Float16 *W1,*W2,*W3,*We,*Wps,*Wpo,*Wpr,*Wase,*Wasm,*Waoe,*Waom,
                 *Warr,*Warm,*Wfs,*Wfo,*Wfr;
};

static inline void gemm(hipStream_t s, const float* A, const _Float16* Wt,
                        const float* bias, const float* resid, float* C, int M) {
  k_gemm256<<<dim3(M / 32), dim3(256), 0, s>>>(A, Wt, bias, resid, C, M);
}

static void run_iter(hipStream_t s, const P& p,
                     const float* sub_in, const float* obj_in, const float* rel_in,
                     float* sub_out, float* obj_out, float* rel_out,
                     float* F, float* T1, float* T2, float* T3, float* T4,
                     float* L1, float* L2, float* HT, float* Q, float* HW,
                     float* CTX, float* EN)
{
  const int EBLK = (ROWS_E + 7) / 8;
  const int RBLK = (ROWS_R + 7) / 8;
  const int NENG = kBS * kNE * (kNE + 1);

  // ---- factor = LN((l1 * l2 * l3) @ We + be) ----
  gemm(s, sub_in, p.W1, p.b1, nullptr, L1, ROWS_E);
  gemm(s, obj_in, p.W2, p.b2, nullptr, L2, ROWS_E);
  gemm(s, rel_in, p.W3, p.b3, nullptr, T1, ROWS_R);          // l3
  k_prod3<<<(int)(SZ_R / 256), 256, 0, s>>>(L1, L2, T1, T2);
  gemm(s, T2, p.We, p.be, nullptr, F, ROWS_R);
  k_ln<<<RBLK, 256, 0, s>>>(F, F, ROWS_R);

  // ---- subjects ----
  gemm(s, sub_in, p.Wps, p.bps, nullptr, HT, ROWS_E);        // ht
  gemm(s, sub_in, p.Wase, p.bas, nullptr, Q, ROWS_E);        // q
  gemm(s, HT, p.Wasm, nullptr, nullptr, HW, ROWS_E);         // self row @ Wasm
  gemm(s, F, p.Wasm, nullptr, nullptr, T1, ROWS_R);          // factor @ Wasm
  k_energy_sub<<<(NENG + 7) / 8, 256, 0, s>>>(Q, HW, T1, p.sv, EN);
  k_softctx_sub<<<ROWS_E, 256, 0, s>>>(EN, HT, F, p.ent, CTX);
  gemm(s, CTX, p.Wfs, p.bfs, sub_in, HW, ROWS_E);            // + residual
  k_ln<<<EBLK, 256, 0, s>>>(HW, sub_out, ROWS_E);

  // ---- objects ----
  gemm(s, obj_in, p.Wpo, p.bpo, nullptr, HT, ROWS_E);
  gemm(s, obj_in, p.Waoe, p.bao, nullptr, Q, ROWS_E);
  gemm(s, HT, p.Waom, nullptr, nullptr, HW, ROWS_E);
  gemm(s, F, p.Waom, nullptr, nullptr, T1, ROWS_R);
  k_energy_obj<<<(NENG + 7) / 8, 256, 0, s>>>(Q, HW, T1, p.ov, EN);
  k_softctx_obj<<<ROWS_E, 256, 0, s>>>(EN, HT, F, p.ent, CTX);
  gemm(s, CTX, p.Wfo, p.bfo, obj_in, HW, ROWS_E);
  k_ln<<<EBLK, 256, 0, s>>>(HW, obj_out, ROWS_E);

  // ---- relations ----
  gemm(s, rel_in, p.Wpr, p.bpr, nullptr, T1, ROWS_R);        // htr
  gemm(s, rel_in, p.Warr, p.bar, nullptr, T2, ROWS_R);       // rel @ Warr + bar
  gemm(s, T1, p.Warm, nullptr, nullptr, T3, ROWS_R);         // g0
  gemm(s, F,  p.Warm, nullptr, nullptr, T4, ROWS_R);         // g1
  k_rel_ctx<<<RBLK, 256, 0, s>>>(T2, T3, T4, T1, F, p.rv, T3); // ctx -> T3
  gemm(s, T3, p.Wfr, p.bfr, rel_in, T2, ROWS_R);
  k_ln<<<RBLK, 256, 0, s>>>(T2, rel_out, ROWS_R);
}

extern "C" void kernel_launch(void* const* d_in, const int* in_sizes, int n_in,
                              void* d_out, int out_size, void* d_ws, size_t ws_size,
                              hipStream_t stream)
{
  (void)in_sizes; (void)n_in; (void)out_size; (void)ws_size;

  const float* sub = (const float*)d_in[0];
  const float* obj = (const float*)d_in[1];
  const float* rel = (const float*)d_in[2];

  // fp32 weight pointers (GEMM weights converted below)
  const float* fW[16] = {
    (const float*)d_in[4],  (const float*)d_in[6],  (const float*)d_in[8],
    (const float*)d_in[10], (const float*)d_in[12], (const float*)d_in[14],
    (const float*)d_in[16], (const float*)d_in[18], (const float*)d_in[19],
    (const float*)d_in[21], (const float*)d_in[22], (const float*)d_in[24],
    (const float*)d_in[25], (const float*)d_in[30], (const float*)d_in[32],
    (const float*)d_in[34]
  }; // W1 W2 W3 We Wps Wpo Wpr Wase Wasm Waoe Waom Warr Warm Wfs Wfo Wfr

  float* ws = (float*)d_ws;
  size_t off = 0;
  auto take = [&](size_t n) { float* r = ws + off; off += n; return r; };
  float* F    = take(SZ_R);
  float* T1   = take(SZ_R);
  float* T2   = take(SZ_R);
  float* T3   = take(SZ_R);
  float* T4   = take(SZ_R);
  float* RELP = take(SZ_R);
  float* SUBP = take(SZ_E);
  float* OBJP = take(SZ_E);
  float* L1   = take(SZ_E);
  float* L2   = take(SZ_E);
  float* HT   = take(SZ_E);
  float* Q    = take(SZ_E);
  float* HW   = take(SZ_E);
  float* CTX  = take(SZ_E);
  float* EN   = take((size_t)kBS * (kNE + 1) * kNE);
  // fp16 transposed weights: 16 x 65536 halves (2 MB) at the tail
  _Float16* hW[16];
  {
    _Float16* hbase = (_Float16*)take((SZ_W * 16 + 1) / 2);
    for (int w = 0; w < 16; ++w) hW[w] = hbase + (size_t)w * SZ_W;
  }

  // one-time weight convert+transpose (runs on `stream`, graph-capture safe)
  for (int w = 0; w < 16; ++w)
    k_cvt_wt<<<dim3((int)(SZ_W / 256)), dim3(256), 0, stream>>>(fW[w], hW[w]);

  P p;
  p.ent = (const int*)d_in[3];
  p.b1  = (const float*)d_in[5];  p.b2  = (const float*)d_in[7];
  p.b3  = (const float*)d_in[9];  p.be  = (const float*)d_in[11];
  p.bps = (const float*)d_in[13]; p.bpo = (const float*)d_in[15];
  p.bpr = (const float*)d_in[17]; p.bas = (const float*)d_in[20];
  p.bao = (const float*)d_in[23]; p.bar = (const float*)d_in[26];
  p.sv  = (const float*)d_in[27]; p.ov  = (const float*)d_in[28];
  p.rv  = (const float*)d_in[29]; p.bfs = (const float*)d_in[31];
  p.bfo = (const float*)d_in[33]; p.bfr = (const float*)d_in[35];
  p.W1 = hW[0];  p.W2 = hW[1];  p.W3 = hW[2];  p.We = hW[3];
  p.Wps = hW[4]; p.Wpo = hW[5]; p.Wpr = hW[6]; p.Wase = hW[7];
  p.Wasm = hW[8]; p.Waoe = hW[9]; p.Waom = hW[10]; p.Warr = hW[11];
  p.Warm = hW[12]; p.Wfs = hW[13]; p.Wfo = hW[14]; p.Wfr = hW[15];

  float* outS = (float*)d_out;
  float* outO = outS + SZ_E;
  float* outR = outO + SZ_E;

  // iteration 0: inputs -> ping buffers; iteration 1: ping -> d_out
  run_iter(stream, p, sub, obj, rel, SUBP, OBJP, RELP,
           F, T1, T2, T3, T4, L1, L2, HT, Q, HW, CTX, EN);
  run_iter(stream, p, SUBP, OBJP, RELP, outS, outO, outR,
           F, T1, T2, T3, T4, L1, L2, HT, Q, HW, CTX, EN);
}
